// SetConv1dDecoder_858993459737
// MI455X (gfx1250) — matrix operand order repaired and checked by reading, not compiled, not run on hardware
//
#include <hip/hip_runtime.h>
#include <hip/hip_bf16.h>

typedef __attribute__((ext_vector_type(16))) _Float16 v16h;
typedef __attribute__((ext_vector_type(8)))  _Float16 v8h;
typedef __attribute__((ext_vector_type(4)))  _Float16 v4h;
typedef __attribute__((ext_vector_type(8)))  float    v8f;

constexpr int B_ = 4;
constexpr int N_ = 4096;
constexpr int M_ = 8192;
constexpr int C_ = 64;
constexpr int TILE_N = 16;
constexpr int CHUNK  = 32;          // K per v_wmma_f32_16x16x32_f16
constexpr int WAVES_PER_BLOCK = 4;
constexpr int NTILES = B_ * (N_ / TILE_N);   // 1024 wave-tiles

// ---------------------------------------------------------------------------
// Pass 1 (optional): convert z [B,C,M] f32 -> f16 into workspace. Saves 64
// f32->f16 converts per lane per chunk in the main loop (z is re-read by 256
// N-tiles per batch; convert it exactly once instead).
// ---------------------------------------------------------------------------
__global__ __launch_bounds__(256) void zconvert_kernel(const float* __restrict__ z,
                                                       _Float16* __restrict__ zh,
                                                       int n4) {
    int i = blockIdx.x * blockDim.x + threadIdx.x;
    if (i < n4) {
        float4 v = ((const float4*)z)[i];
        v4h h;
        h[0] = (_Float16)v.x; h[1] = (_Float16)v.y;
        h[2] = (_Float16)v.z; h[3] = (_Float16)v.w;
        ((v4h*)zh)[i] = h;
    }
}

// ---------------------------------------------------------------------------
// Main kernel: one wave -> 16(N) x 64(C) output tile, accumulating over M in
// chunks of 32 with v_wmma_f32_16x16x32_f16. Weights (A matrix) are generated
// on the fly with v_exp_f32 (exp2 with log2(e) folded into the scale factor)
// and packed to f16 in contiguous k order (k = 16*half + j), as B is.
// ---------------------------------------------------------------------------
template <bool F16Z>
__global__ __launch_bounds__(WAVES_PER_BLOCK * 32)
void setconv_wmma_kernel(const float*    __restrict__ xz,
                         const float*    __restrict__ zf,
                         const _Float16* __restrict__ zh,
                         const float*    __restrict__ x,
                         const float*    __restrict__ lsp,
                         float*          __restrict__ out) {
    const int lane = threadIdx.x & 31;
    const int wave = threadIdx.x >> 5;
    const int tile = blockIdx.x * WAVES_PER_BLOCK + wave;   // 0 .. NTILES-1
    const int b     = tile / (N_ / TILE_N);
    const int nbase = (tile % (N_ / TILE_N)) * TILE_N;

    const int half = lane >> 4;    // 0: lanes 0-15, 1: lanes 16-31
    const int col  = lane & 15;

    // w = exp(-0.5*d^2/exp(2*ls)) = exp2(coef2 * d^2),
    // coef2 = -0.5*log2(e)*exp(-2*ls)  -> feeds v_exp_f32 directly
    const float ls    = lsp[0];
    const float coef2 = -0.72134752044448170f * __expf(-2.0f * ls);

    // A-matrix row value for this lane (row = lane & 15)
    const float xq = x[(size_t)b * N_ + nbase + col];

    // xz base including the per-half K offset of this code's contiguous A order:
    // lanes 0-15 hold K = 0..15; lanes 16-31 hold K = 16..31
    const float* xzb = xz + (size_t)b * M_ + half * 16;

    // B-matrix: lane holds column c = t*16 + col; K = 16*half + 0..15
    // (contiguous along z's M axis -> b128 loads)
    size_t zoff[4];
#pragma unroll
    for (int t = 0; t < 4; ++t)
        zoff[t] = ((size_t)b * C_ + t * 16 + col) * (size_t)M_ + half * 16;

    v8f acc[4] = {};   // 4 C-tiles of the 16x16 f32 accumulator

#pragma unroll 2
    for (int mb = 0; mb < M_; mb += CHUNK) {
        // ---- A fragment: 16 RBF weights per lane --------------------------
        const float4 xa0 = *(const float4*)(xzb + mb);
        const float4 xa1 = *(const float4*)(xzb + mb + 4);
        const float4 xb0 = *(const float4*)(xzb + mb + 8);
        const float4 xb1 = *(const float4*)(xzb + mb + 12);
        const float mv[16] = { xa0.x, xa0.y, xa0.z, xa0.w,
                               xa1.x, xa1.y, xa1.z, xa1.w,
                               xb0.x, xb0.y, xb0.z, xb0.w,
                               xb1.x, xb1.y, xb1.z, xb1.w };
        v16h a;
#pragma unroll
        for (int j = 0; j < 16; ++j) {
            const float d  = xq - mv[j];
            const float cd = coef2 * d;
            a[j] = (_Float16)__builtin_amdgcn_exp2f(cd * d);  // v_mul, v_mul, v_exp_f32
        }

        // ---- 4 B fragments + 4 WMMAs --------------------------------------
#pragma unroll
        for (int t = 0; t < 4; ++t) {
            v16h bf;
            if (F16Z) {
                const v8h p0 = *(const v8h*)(zh + zoff[t] + mb);
                const v8h p1 = *(const v8h*)(zh + zoff[t] + mb + 8);
#pragma unroll
                for (int j = 0; j < 8; ++j) { bf[j] = p0[j]; bf[j + 8] = p1[j]; }
            } else {
                const float4 z0 = *(const float4*)(zf + zoff[t] + mb);
                const float4 z1 = *(const float4*)(zf + zoff[t] + mb + 4);
                const float4 z2 = *(const float4*)(zf + zoff[t] + mb + 8);
                const float4 z3 = *(const float4*)(zf + zoff[t] + mb + 12);
                const float zv[16] = { z0.x, z0.y, z0.z, z0.w,
                                       z1.x, z1.y, z1.z, z1.w,
                                       z2.x, z2.y, z2.z, z2.w,
                                       z3.x, z3.y, z3.z, z3.w };
#pragma unroll
                for (int j = 0; j < 16; ++j) bf[j] = (_Float16)zv[j];
            }
            // D = A(16x32 f16) x B(32x16 f16) + C(16x16 f32)
            acc[t] = __builtin_amdgcn_wmma_f32_16x16x32_f16(
                false, a, false, bf, (short)0, acc[t], false, false);
        }
    }

    // ---- store: D VGPR r <-> row (r + 8*half), col = lane&15 --------------
    float* orow = out + ((size_t)b * N_ + nbase + half * 8) * C_ + col;
#pragma unroll
    for (int t = 0; t < 4; ++t) {
#pragma unroll
        for (int r = 0; r < 8; ++r)
            orow[(size_t)r * C_ + t * 16] = acc[t][r];
    }
}

extern "C" void kernel_launch(void* const* d_in, const int* in_sizes, int n_in,
                              void* d_out, int out_size, void* d_ws, size_t ws_size,
                              hipStream_t stream) {
    const float* xz = (const float*)d_in[0];   // [B, M, 1]
    const float* z  = (const float*)d_in[1];   // [B, C, M]
    const float* x  = (const float*)d_in[2];   // [B, N, 1]
    const float* ls = (const float*)d_in[3];   // scalar
    float* out = (float*)d_out;                // [B, N, C]

    const dim3 grid(NTILES / WAVES_PER_BLOCK);
    const dim3 block(WAVES_PER_BLOCK * 32);

    const size_t zh_bytes = (size_t)B_ * C_ * M_ * 2;   // 4 MiB
    if (ws_size >= zh_bytes) {
        _Float16* zh = (_Float16*)d_ws;
        const int n4 = B_ * C_ * M_ / 4;
        zconvert_kernel<<<(n4 + 255) / 256, 256, 0, stream>>>(z, zh, n4);
        setconv_wmma_kernel<true><<<grid, block, 0, stream>>>(xz, z, zh, x, ls, out);
    } else {
        setconv_wmma_kernel<false><<<grid, block, 0, stream>>>(xz, z, (const _Float16*)nullptr, x, ls, out);
    }
}
